// TransformerAttentionBlock_62646392979837
// MI455X (gfx1250) — matrix operand order repaired and checked by reading, not compiled, not run on hardware
//
#include <hip/hip_runtime.h>
#include <hip/hip_bf16.h>

// ---------------- problem constants ----------------
#define BATCH   2
#define CDIM    512
#define SIDE    48
#define NTOK    (SIDE * SIDE)          // 2304
#define HEADS   8
#define DHEAD   64
#define INNER   (HEADS * DHEAD)        // 512
#define MROWS   (BATCH * NTOK)         // 4608
#define EPSLN   1e-5f

typedef __attribute__((ext_vector_type(16))) _Float16 v16h;
typedef __attribute__((ext_vector_type(8)))  float    v8f;

// ---------------- async global->LDS (CDNA5 data mover path) ----------------
#if defined(__gfx1250__) &&                                                   \
    __has_builtin(__builtin_amdgcn_global_load_async_to_lds_b128) &&          \
    __has_builtin(__builtin_amdgcn_s_wait_asynccnt)
#define USE_ASYNC 1
typedef int i128v __attribute__((vector_size(16)));
typedef __attribute__((address_space(1))) i128v* gptr128;   // global int4*
typedef __attribute__((address_space(3))) i128v* lptr128;   // LDS int4*
__device__ __forceinline__ void g2l_b128(void* lds, const void* g) {
  __builtin_amdgcn_global_load_async_to_lds_b128(
      (gptr128)(uintptr_t)g, (lptr128)lds, 0, 0);
}
__device__ __forceinline__ void wait_async() {
  __builtin_amdgcn_s_wait_asynccnt(0);
}
#else
__device__ __forceinline__ void g2l_b128(void* lds, const void* g) {
  *(uint4*)lds = *(const uint4*)g;
}
__device__ __forceinline__ void wait_async() {}
#endif

// ---------------- cross-lane reductions (DPP, no LDS) ----------------
#if defined(__gfx1250__) && __has_builtin(__builtin_amdgcn_update_dpp)
template <int CTRL>
__device__ __forceinline__ float dppf(float x) {
  return __builtin_bit_cast(float,
      __builtin_amdgcn_update_dpp(0, __builtin_bit_cast(int, x),
                                  CTRL, 0xF, 0xF, true));
}
// full reduction across each 16-lane half (all lanes get the result)
__device__ __forceinline__ float redmax16(float x) {
  x = fmaxf(x, dppf<0xB1>(x));    // quad_perm(1,0,3,2): ^1
  x = fmaxf(x, dppf<0x4E>(x));    // quad_perm(2,3,0,1): ^2
  x = fmaxf(x, dppf<0x141>(x));   // row_half_mirror:    ^7
  x = fmaxf(x, dppf<0x140>(x));   // row_mirror:         ^15
  return x;
}
__device__ __forceinline__ float redsum16(float x) {
  x += dppf<0xB1>(x);
  x += dppf<0x4E>(x);
  x += dppf<0x141>(x);
  x += dppf<0x140>(x);
  return x;
}
#else
__device__ __forceinline__ float redmax16(float x) {
#pragma unroll
  for (int m = 1; m < 16; m <<= 1) x = fmaxf(x, __shfl_xor(x, m, 32));
  return x;
}
__device__ __forceinline__ float redsum16(float x) {
#pragma unroll
  for (int m = 1; m < 16; m <<= 1) x += __shfl_xor(x, m, 32);
  return x;
}
#endif

#if defined(__gfx1250__) && __has_builtin(__builtin_amdgcn_permlanex16)
__device__ __forceinline__ float xhalf32(float x) {   // swap 16-lane halves
  int v = __builtin_amdgcn_permlanex16(
      __builtin_bit_cast(int, x), __builtin_bit_cast(int, x),
      0x76543210, 0xfedcba98, false, false);
  return __builtin_bit_cast(float, v);
}
#else
__device__ __forceinline__ float xhalf32(float x) { return __shfl_xor(x, 16, 32); }
#endif

// ---------------- WMMA helpers ----------------
__device__ __forceinline__ v8f wmma_f16(v16h a, v16h b, v8f c) {
  return __builtin_amdgcn_wmma_f32_16x16x32_f16(false, a, false, b,
                                                (short)0, c, false, false);
}

// A operand, logical A[m][k] (m<16, k<32), src row-major, row stride ld halves.
__device__ __forceinline__ v16h load_a16x32(const _Float16* src, int ld, int lane) {
  const _Float16* r = src + (lane & 15) * ld + (lane >> 4) * 8;
  v16h a;
  *(uint4*)&a       = *(const uint4*)r;         // k = bk .. bk+7
  *((uint4*)&a + 1) = *(const uint4*)(r + 16);  // k = 16+bk .. 16+bk+7
  return a;
}

// B operand, logical B[k][n] (k<32, n<16), memory k-contiguous at fixed n:
// B[k][n] = src[n*ld + k].
__device__ __forceinline__ v16h load_brun(const _Float16* src, int ld, int lane) {
  const _Float16* r = src + (lane & 15) * ld + (lane >> 4) * 8;
  v16h b;
  *(uint4*)&b       = *(const uint4*)r;
  *((uint4*)&b + 1) = *(const uint4*)(r + 16);
  return b;
}

// ---------------- weight convert: f32 KxN -> f16 NxK (transposed) ----------
__global__ void k_w_to_f16_t(const float* __restrict__ s, _Float16* __restrict__ d,
                             int K, int N) {
  int o = blockIdx.x * 256 + threadIdx.x;
  if (o < K * N) {
    int n = o / K, k = o - n * K;
    d[o] = (_Float16)s[(size_t)k * N + n];
  }
}

// ---------------- block reduction (128 threads = 4 waves) ----------------
__device__ __forceinline__ float blockSum4(float v, volatile float* sb) {
  int tid = threadIdx.x;
  v = redsum16(v);
  v += xhalf32(v);
  if ((tid & 31) == 0) sb[tid >> 5] = v;
  __syncthreads();
  float s = sb[0] + sb[1] + sb[2] + sb[3];
  __syncthreads();
  return s;
}

// ---------------- LN1: transpose (B,C,n)->(B,n,C), LN over C ---------------
__global__ void k_ln1(const float* __restrict__ x, const float* __restrict__ g,
                      const float* __restrict__ bb, float* __restrict__ xs,
                      _Float16* __restrict__ xln) {
  __shared__ float sb[4];
  int bt = blockIdx.x;
  int b = bt / NTOK, t = bt - b * NTOK;
  const float* xb = x + (size_t)b * CDIM * NTOK + t;
  float v[4];
#pragma unroll
  for (int j = 0; j < 4; ++j) v[j] = xb[(size_t)(threadIdx.x + j * 128) * NTOK];
  float mean = blockSum4(v[0] + v[1] + v[2] + v[3], sb) * (1.f / CDIM);
  float d2 = 0.f;
#pragma unroll
  for (int j = 0; j < 4; ++j) { float d = v[j] - mean; d2 += d * d; }
  float rstd = rsqrtf(blockSum4(d2, sb) * (1.f / CDIM) + EPSLN);
  float* xso = xs + (size_t)bt * CDIM;
  _Float16* xlo = xln + (size_t)bt * CDIM;
#pragma unroll
  for (int j = 0; j < 4; ++j) {
    int c = threadIdx.x + j * 128;
    xso[c] = v[j];
    xlo[c] = (_Float16)((v[j] - mean) * rstd * g[c] + bb[c]);
  }
}

// ---------------- LN(a + xs) -> fp32 ----------------
__global__ void k_ln_res(const float* __restrict__ a, const float* __restrict__ xs,
                         const float* __restrict__ g, const float* __restrict__ bb,
                         float* __restrict__ av) {
  __shared__ float sb[4];
  size_t base = (size_t)blockIdx.x * CDIM;
  float v[4];
#pragma unroll
  for (int j = 0; j < 4; ++j) {
    int c = threadIdx.x + j * 128;
    v[j] = a[base + c] + xs[base + c];
  }
  float mean = blockSum4(v[0] + v[1] + v[2] + v[3], sb) * (1.f / CDIM);
  float d2 = 0.f;
#pragma unroll
  for (int j = 0; j < 4; ++j) { float d = v[j] - mean; d2 += d * d; }
  float rstd = rsqrtf(blockSum4(d2, sb) * (1.f / CDIM) + EPSLN);
#pragma unroll
  for (int j = 0; j < 4; ++j) {
    int c = threadIdx.x + j * 128;
    av[base + c] = (v[j] - mean) * rstd * g[c] + bb[c];
  }
}

// ---------------- LN -> f16 (optionally exact GELU) ----------------
template <bool GELU>
__global__ void k_ln16(const float* __restrict__ a, const float* __restrict__ g,
                       const float* __restrict__ bb, _Float16* __restrict__ o) {
  __shared__ float sb[4];
  size_t base = (size_t)blockIdx.x * CDIM;
  float v[4];
#pragma unroll
  for (int j = 0; j < 4; ++j) v[j] = a[base + threadIdx.x + j * 128];
  float mean = blockSum4(v[0] + v[1] + v[2] + v[3], sb) * (1.f / CDIM);
  float d2 = 0.f;
#pragma unroll
  for (int j = 0; j < 4; ++j) { float d = v[j] - mean; d2 += d * d; }
  float rstd = rsqrtf(blockSum4(d2, sb) * (1.f / CDIM) + EPSLN);
#pragma unroll
  for (int j = 0; j < 4; ++j) {
    int c = threadIdx.x + j * 128;
    float u = (v[j] - mean) * rstd * g[c] + bb[c];
    if (GELU) u = 0.5f * u * (1.f + erff(u * 0.70710678118f));
    o[base + c] = (_Float16)u;
  }
}

// ---------------- f16 WMMA GEMM:  C[M,N] = scale*(A @ Bt^T) + bias ---------
// A: M x K row-major f16.  Bt: N x K row-major f16 (pre-transposed weight).
// 256 threads = 8 waves; block tile 128x64; wave tile 32x32; K-step 32.
// Double-buffered LDS + async loads: one barrier per K-step.
template <bool BIAS, bool F32OUT, bool F16OUT>
__global__ void k_gemm(const _Float16* __restrict__ A, const _Float16* __restrict__ Bt,
                       const float* __restrict__ bias, float* __restrict__ Cf,
                       _Float16* __restrict__ Ch, int M, int N, int K, float scale) {
  __shared__ __align__(16) _Float16 As[2][128 * 32];
  __shared__ __align__(16) _Float16 BsT[2][64 * 32];
  int tid = threadIdx.x, lane = tid & 31;
  int wave = __builtin_amdgcn_readfirstlane(tid >> 5);
  int m0 = blockIdx.x * 128, n0 = blockIdx.y * 64;
  int rw = wave & 3, cw = wave >> 2;   // 4 row-groups x 2 col-groups of 32
  v8f a00 = {}, a01 = {}, a10 = {}, a11 = {};

  int au0 = tid * 2, au1 = tid * 2 + 1;               // 512 uint4 for As
  int ar0 = au0 >> 2, as0 = (au0 & 3) * 8;
  int ar1 = au1 >> 2, as1 = (au1 & 3) * 8;
  int bn = tid >> 2, bs = (tid & 3) * 8;              // 256 uint4 for BsT

  auto issue = [&](int buf, int k0) {
    g2l_b128(&As[buf][ar0 * 32 + as0], &A[(size_t)(m0 + ar0) * K + k0 + as0]);
    g2l_b128(&As[buf][ar1 * 32 + as1], &A[(size_t)(m0 + ar1) * K + k0 + as1]);
    g2l_b128(&BsT[buf][bn * 32 + bs], &Bt[(size_t)(n0 + bn) * K + k0 + bs]);
    if (k0 + 32 < K)
      __builtin_prefetch(&A[(size_t)(m0 + ar0) * K + k0 + 32 + as0], 0, 1);
  };

  int nk = K >> 5;
  issue(0, 0);
  for (int i = 0; i < nk; ++i) {
    int cur = i & 1;
    wait_async();            // this wave's loads for buf[cur] complete
    __syncthreads();         // everyone's loads visible; prev reads of nxt done
    if (i + 1 < nk) issue(cur ^ 1, (i + 1) * 32);
    const _Float16* as = As[cur];
    const _Float16* bst = BsT[cur];
    v16h a0 = load_a16x32(as + (rw * 32) * 32, 32, lane);
    v16h a1 = load_a16x32(as + (rw * 32 + 16) * 32, 32, lane);
    v16h b0 = load_brun(bst + (cw * 32) * 32, 32, lane);
    v16h b1 = load_brun(bst + (cw * 32 + 16) * 32, 32, lane);
    a00 = wmma_f16(a0, b0, a00);
    a01 = wmma_f16(a0, b1, a01);
    a10 = wmma_f16(a1, b0, a10);
    a11 = wmma_f16(a1, b1, a11);
  }

  int hi = lane >> 4, col = lane & 15;
  float bi0 = BIAS ? bias[n0 + cw * 32 + col] : 0.f;
  float bi1 = BIAS ? bias[n0 + cw * 32 + 16 + col] : 0.f;
  const v8f* accs[4] = {&a00, &a01, &a10, &a11};
#pragma unroll
  for (int ti = 0; ti < 2; ++ti)
#pragma unroll
    for (int tj = 0; tj < 2; ++tj) {
      const v8f& ac = *accs[ti * 2 + tj];
      float bb = tj ? bi1 : bi0;
#pragma unroll
      for (int r = 0; r < 8; ++r) {
        int row = m0 + rw * 32 + ti * 16 + r + 8 * hi;
        size_t idx = (size_t)row * N + n0 + cw * 32 + tj * 16 + col;
        float v = ac[r] * scale + bb;
        if (F32OUT) Cf[idx] = v;
        if (F16OUT) Ch[idx] = (_Float16)v;
      }
    }
}

// ---------------- causal flash attention ----------------
// grid (18, HEADS, BATCH); 8 waves; one 16-row query tile per wave;
// 32-key chunks double-buffered in LDS, shared by all waves.
__global__ void k_attn(const _Float16* __restrict__ qh, const _Float16* __restrict__ kvh,
                       _Float16* __restrict__ aout) {
  __shared__ __align__(16) _Float16 Kt[2][32 * 64];   // [key][d]
  __shared__ __align__(16) _Float16 VtT[2][64 * 32];  // [d][key]
  __shared__ __align__(16) _Float16 Pl[8][16 * 32];
  int b = blockIdx.z, h = blockIdx.y, bq = blockIdx.x;
  int tid = threadIdx.x, lane = tid & 31;
  int wave = __builtin_amdgcn_readfirstlane(tid >> 5);
  int hi = lane >> 4, col = lane & 15;
  int q0 = (bq * 8 + wave) * 16;

  const _Float16* qbase = qh + ((size_t)(b * NTOK + q0)) * INNER + h * DHEAD;
  v16h qa0 = load_a16x32(qbase, INNER, lane);
  v16h qa1 = load_a16x32(qbase + 32, INNER, lane);

  v8f acc[4] = {};
  float mrow[8], lrow[8];
#pragma unroll
  for (int r = 0; r < 8; ++r) { mrow[r] = -3.0e38f; lrow[r] = 0.f; }

  int ldrow = tid >> 3, ldseg = (tid & 7) * 8;
  auto issueKV = [&](int buf, int j0) {
    const _Float16* src = kvh + ((size_t)(b * NTOK + j0 + ldrow)) * (2 * DHEAD);
    g2l_b128(&Kt[buf][ldrow * 64 + ldseg], src + ldseg);   // K row-major (async)
    uint4 vv = *(const uint4*)(src + DHEAD + ldseg);       // V transpose
    const _Float16* vh = (const _Float16*)&vv;
#pragma unroll
    for (int j = 0; j < 8; ++j) VtT[buf][(ldseg + j) * 32 + ldrow] = vh[j];
  };

  int nchunks = bq * 4 + 4;
  issueKV(0, 0);
  for (int c = 0; c < nchunks; ++c) {
    int j0 = c * 32;
    int cur = c & 1;
    wait_async();
    __syncthreads();
    if (c + 1 < nchunks) issueKV(cur ^ 1, (c + 1) * 32);
    if (j0 <= q0 + 15) {                       // chunk intersects causal region
      const _Float16* kt = Kt[cur];
      const _Float16* vt = VtT[cur];
      v8f s0 = {}, s1 = {};
      s0 = wmma_f16(qa0, load_brun(kt, 64, lane), s0);
      s0 = wmma_f16(qa1, load_brun(kt + 32, 64, lane), s0);
      s1 = wmma_f16(qa0, load_brun(kt + 16 * 64, 64, lane), s1);
      s1 = wmma_f16(qa1, load_brun(kt + 16 * 64 + 32, 64, lane), s1);

#pragma unroll
      for (int r = 0; r < 8; ++r) {
        int ro = q0 + r + 8 * hi;
        float v0 = (j0 + col      <= ro) ? s0[r] : -3.0e38f;
        float v1 = (j0 + 16 + col <= ro) ? s1[r] : -3.0e38f;
        float mnew = fmaxf(mrow[r], redmax16(fmaxf(v0, v1)));
        float corr = __expf(mrow[r] - mnew);
        float p0 = __expf(v0 - mnew);
        float p1 = __expf(v1 - mnew);
        s0[r] = p0; s1[r] = p1;
        lrow[r] = lrow[r] * corr + redsum16(p0 + p1);
        mrow[r] = mnew;
#pragma unroll
        for (int t = 0; t < 4; ++t) acc[t][r] *= corr;
      }

      // P: D-layout -> A-layout via per-wave LDS scratch
      _Float16* pl = &Pl[wave][0];
#pragma unroll
      for (int r = 0; r < 8; ++r) {
        int rr = r + 8 * hi;
        pl[rr * 32 + col]      = (_Float16)s0[r];
        pl[rr * 32 + 16 + col] = (_Float16)s1[r];
      }
      v16h pa = load_a16x32(pl, 32, lane);
#pragma unroll
      for (int t = 0; t < 4; ++t)
        acc[t] = wmma_f16(pa, load_brun(vt + (t * 16) * 32, 32, lane), acc[t]);
    }
  }

  float inv[8];
#pragma unroll
  for (int r = 0; r < 8; ++r) inv[r] = 1.0f / lrow[r];
#pragma unroll
  for (int t = 0; t < 4; ++t)
#pragma unroll
    for (int r = 0; r < 8; ++r) {
      int ro = q0 + r + 8 * hi;
      float o = acc[t][r] * inv[r];
      aout[((size_t)(b * NTOK + ro)) * INNER + h * DHEAD + t * 16 + col] = (_Float16)o;
    }
}

// ---------------- final: residual + transpose back to (B,C,n) -------------
__global__ void k_final(const float* __restrict__ ff2, const float* __restrict__ av,
                        float* __restrict__ y) {
  size_t i = (size_t)blockIdx.x * 256 + threadIdx.x;
  size_t row = i / CDIM;
  int c = (int)(i - row * CDIM);
  int b = (int)(row / NTOK), t = (int)(row - (size_t)b * NTOK);
  y[(size_t)b * CDIM * NTOK + (size_t)c * NTOK + t] = ff2[i] + av[i];
}

// ---------------- host launch ----------------
extern "C" void kernel_launch(void* const* d_in, const int* in_sizes, int n_in,
                              void* d_out, int out_size, void* d_ws, size_t ws_size,
                              hipStream_t stream) {
  const float* x      = (const float*)d_in[0];
  const float* ln1_g  = (const float*)d_in[1];
  const float* ln1_b  = (const float*)d_in[2];
  const float* wq     = (const float*)d_in[3];
  const float* wkv    = (const float*)d_in[4];
  const float* wo     = (const float*)d_in[5];
  const float* ln2_g  = (const float*)d_in[6];
  const float* ln2_b  = (const float*)d_in[7];
  const float* ffl1_g = (const float*)d_in[8];
  const float* ffl1_b = (const float*)d_in[9];
  const float* ff_w1  = (const float*)d_in[10];
  const float* ff_b1  = (const float*)d_in[11];
  const float* ffl2_g = (const float*)d_in[12];
  const float* ffl2_b = (const float*)d_in[13];
  const float* ff_w2  = (const float*)d_in[14];
  const float* ff_b2  = (const float*)d_in[15];
  float* y            = (float*)d_out;

  char* p = (char*)d_ws;
  auto carve = [&](size_t bytes) {
    void* r = (void*)p;
    p += (bytes + 255) & ~(size_t)255;
    return r;
  };
  const size_t MC = (size_t)MROWS * CDIM;
  float*    xs     = (float*)    carve(MC * 4);
  _Float16* xln    = (_Float16*) carve(MC * 2);
  _Float16* q_h    = (_Float16*) carve(MC * 2);
  _Float16* kv_h   = (_Float16*) carve((size_t)MROWS * 128 * 2);
  _Float16* attn_h = (_Float16*) carve(MC * 2);
  float*    proj   = (float*)    carve(MC * 4);
  float*    av     = (float*)    carve(MC * 4);
  _Float16* ffl1h  = (_Float16*) carve(MC * 2);
  float*    ff1    = (float*)    carve(MC * 4);
  _Float16* ffgh   = (_Float16*) carve(MC * 2);
  float*    ff2    = (float*)    carve(MC * 4);
  _Float16* wqT    = (_Float16*) carve((size_t)CDIM * INNER * 2);   // INNER x CDIM
  _Float16* wkvT   = (_Float16*) carve((size_t)CDIM * 128 * 2);     // 128 x CDIM
  _Float16* woT    = (_Float16*) carve((size_t)INNER * CDIM * 2);   // CDIM x INNER
  _Float16* w1T    = (_Float16*) carve((size_t)CDIM * CDIM * 2);
  _Float16* w2T    = (_Float16*) carve((size_t)CDIM * CDIM * 2);

  // 0) convert + transpose weights (K x N -> N x K)
  k_w_to_f16_t<<<(CDIM * INNER + 255) / 256, 256, 0, stream>>>(wq, wqT, CDIM, INNER);
  k_w_to_f16_t<<<(CDIM * 128 + 255) / 256, 256, 0, stream>>>(wkv, wkvT, CDIM, 128);
  k_w_to_f16_t<<<(INNER * CDIM + 255) / 256, 256, 0, stream>>>(wo, woT, INNER, CDIM);
  k_w_to_f16_t<<<(CDIM * CDIM + 255) / 256, 256, 0, stream>>>(ff_w1, w1T, CDIM, CDIM);
  k_w_to_f16_t<<<(CDIM * CDIM + 255) / 256, 256, 0, stream>>>(ff_w2, w2T, CDIM, CDIM);

  // 1) LN1 + transpose
  k_ln1<<<MROWS, 128, 0, stream>>>(x, ln1_g, ln1_b, xs, xln);

  // 2) q (scaled by d^-1/2) and kv projections
  k_gemm<false, false, true><<<dim3(MROWS / 128, INNER / 64), 256, 0, stream>>>(
      xln, wqT, nullptr, nullptr, q_h, MROWS, INNER, CDIM, 0.125f);
  k_gemm<false, false, true><<<dim3(MROWS / 128, 128 / 64), 256, 0, stream>>>(
      xln, wkvT, nullptr, nullptr, kv_h, MROWS, 128, CDIM, 1.0f);

  // 3) causal flash attention
  k_attn<<<dim3(NTOK / 128, HEADS, BATCH), 256, 0, stream>>>(q_h, kv_h, attn_h);

  // 4) output projection
  k_gemm<false, true, false><<<dim3(MROWS / 128, CDIM / 64), 256, 0, stream>>>(
      attn_h, woT, nullptr, proj, nullptr, MROWS, CDIM, INNER, 1.0f);

  // 5) av = LN2(proj + xs)
  k_ln_res<<<MROWS, 128, 0, stream>>>(proj, xs, ln2_g, ln2_b, av);

  // 6) FF: LN -> w1+b1 -> LN -> gelu -> w2+b2
  k_ln16<false><<<MROWS, 128, 0, stream>>>(av, ffl1_g, ffl1_b, ffl1h);
  k_gemm<true, true, false><<<dim3(MROWS / 128, CDIM / 64), 256, 0, stream>>>(
      ffl1h, w1T, ff_b1, ff1, nullptr, MROWS, CDIM, CDIM, 1.0f);
  k_ln16<true><<<MROWS, 128, 0, stream>>>(ff1, ffl2_g, ffl2_b, ffgh);
  k_gemm<true, true, false><<<dim3(MROWS / 128, CDIM / 64), 256, 0, stream>>>(
      ffgh, w2T, ff_b2, ff2, nullptr, MROWS, CDIM, CDIM, 1.0f);

  // 7) y = ff2 + av, transposed back to (B, C, H, W)
  k_final<<<(int)(MC / 256), 256, 0, stream>>>(ff2, av, y);
}
